// LigerCE_74354473828838
// MI455X (gfx1250) — matrix operand order repaired and checked by reading, not compiled, not run on hardware
//
#include <hip/hip_runtime.h>
#include <hip/hip_bf16.h>
#include <math.h>

// Problem constants (from reference)
#define N_ROWS 2048
#define H_DIM  4096
#define V_DIM  32000
#define IGNORE_INDEX (-100)

// Tiling: block tile 64(N) x 128(V), 8 waves as 2x4, each wave owns 2x2 16x16 tiles
#define BN 64
#define BV 128
#define KB 32          // K per step (one bf16 WMMA K)
#define STRIDE 40      // LDS row stride in bf16 elems (80B, 16B aligned, padded)

typedef float          v8f  __attribute__((ext_vector_type(8)));
typedef unsigned int   v8u  __attribute__((ext_vector_type(8)));
typedef __bf16         v16bf __attribute__((ext_vector_type(16)));

// Split fp32 into truncated-bf16 hi + bf16(residual) lo.  hi+lo ~ x to ~2^-16 rel.
__device__ __forceinline__ void split_store4(float4 v, unsigned short* hi, unsigned short* lo) {
    float x[4] = {v.x, v.y, v.z, v.w};
#pragma unroll
    for (int i = 0; i < 4; ++i) {
        unsigned u = __float_as_uint(x[i]);
        unsigned short h = (unsigned short)(u >> 16);           // RTZ to bf16
        float r = x[i] - __uint_as_float(((unsigned)h) << 16);  // exact residual
        hi[i] = h;
        lo[i] = (unsigned short)(__float_as_uint(r) >> 16);
    }
}

// Build a v16bf fragment from two 16B LDS segments (8 bf16 each).
__device__ __forceinline__ v16bf load_frag(const unsigned short* p0, const unsigned short* p1) {
    uint4 a = *(const uint4*)p0;
    uint4 b = *(const uint4*)p1;
    v8u r;
    r[0]=a.x; r[1]=a.y; r[2]=a.z; r[3]=a.w;
    r[4]=b.x; r[5]=b.y; r[6]=b.z; r[7]=b.w;
    return __builtin_bit_cast(v16bf, r);
}

// Fast softcap: 30*tanh(x/30) via hardware exp + rcp (v_exp_f32, v_rcp_f32).
__device__ __forceinline__ float softcap30(float x) {
    float ax = fabsf(x) * (2.0f / 30.0f);
    float t  = __expf(-ax);                       // e^(-2|x|/30)
    float th = (1.0f - t) * __builtin_amdgcn_rcpf(1.0f + t);
    return copysignf(30.0f * th, x);
}

__global__ __launch_bounds__(256)
void lce_gemm_kernel(const float* __restrict__ inp,     // [N,H]
                     const float* __restrict__ wgt,     // [V,H]
                     const float* __restrict__ bias,    // [V]
                     const int*   __restrict__ target,  // [N]
                     float* __restrict__ g_sumexp,      // [N] (atomic accum)
                     float* __restrict__ g_sumlog,      // [N] (atomic accum)
                     float* __restrict__ g_zy)          // [N] (unique writer)
{
    __shared__ unsigned short sAhi[2][BN * STRIDE];
    __shared__ unsigned short sAlo[2][BN * STRIDE];
    __shared__ unsigned short sBhi[2][BV * STRIDE];
    __shared__ unsigned short sBlo[2][BV * STRIDE];
    __shared__ float sExp[BN];
    __shared__ float sSum[BN];
    __shared__ int   sTgt[BN];

    const int tid  = threadIdx.x;
    const int n0   = blockIdx.x * BN;   // x fastest -> 32 N-blocks reuse same weight slab (L2)
    const int v0   = blockIdx.y * BV;
    const int wave = tid >> 5;
    const int lane = tid & 31;
    const int wm   = wave & 1;          // 2 waves along N (2*32 = 64)
    const int wv   = wave >> 1;         // 4 waves along V (4*32 = 128)
    const int half = lane >> 4;
    const int l16  = lane & 15;

    if (tid < BN) {
        int t = target[n0 + tid];
        sTgt[tid] = (t == IGNORE_INDEX) ? 0 : t;
        sExp[tid] = 0.0f;
        sSum[tid] = 0.0f;
    }

    v8f acc[2][2] = {};                 // [tm][tv]

    // Staging: A 64x32 = 2 float4/thread, B 128x32 = 4 float4/thread
    const int ldr = tid >> 3;           // 0..31
    const int ldc = (tid & 7) << 2;     // 0,4,...,28
    const float* aBase = inp + (size_t)(n0 + ldr) * H_DIM + ldc;
    const float* bBase = wgt + (size_t)(v0 + ldr) * H_DIM + ldc;

    // Prefetch registers for next K slab
    float4 pa0 = *(const float4*)(aBase);
    float4 pa1 = *(const float4*)(aBase + (size_t)32 * H_DIM);
    float4 pb0 = *(const float4*)(bBase);
    float4 pb1 = *(const float4*)(bBase + (size_t)32 * H_DIM);
    float4 pb2 = *(const float4*)(bBase + (size_t)64 * H_DIM);
    float4 pb3 = *(const float4*)(bBase + (size_t)96 * H_DIM);

    // Stage K=0 into buffer 0
    split_store4(pa0, &sAhi[0][ ldr       * STRIDE + ldc], &sAlo[0][ ldr       * STRIDE + ldc]);
    split_store4(pa1, &sAhi[0][(ldr + 32) * STRIDE + ldc], &sAlo[0][(ldr + 32) * STRIDE + ldc]);
    split_store4(pb0, &sBhi[0][ ldr       * STRIDE + ldc], &sBlo[0][ ldr       * STRIDE + ldc]);
    split_store4(pb1, &sBhi[0][(ldr + 32) * STRIDE + ldc], &sBlo[0][(ldr + 32) * STRIDE + ldc]);
    split_store4(pb2, &sBhi[0][(ldr + 64) * STRIDE + ldc], &sBlo[0][(ldr + 64) * STRIDE + ldc]);
    split_store4(pb3, &sBhi[0][(ldr + 96) * STRIDE + ldc], &sBlo[0][(ldr + 96) * STRIDE + ldc]);

    int cur = 0;
    for (int k0 = 0; k0 < H_DIM; k0 += KB) {
        // Issue global prefetch of next K slab early (overlaps with WMMA below)
        const int kn = (k0 + KB < H_DIM) ? (k0 + KB) : 0;   // last-iter dummy (in-bounds)
        pa0 = *(const float4*)(aBase + kn);
        pa1 = *(const float4*)(aBase + kn + (size_t)32 * H_DIM);
        pb0 = *(const float4*)(bBase + kn);
        pb1 = *(const float4*)(bBase + kn + (size_t)32 * H_DIM);
        pb2 = *(const float4*)(bBase + kn + (size_t)64 * H_DIM);
        pb3 = *(const float4*)(bBase + kn + (size_t)96 * H_DIM);

        __syncthreads();   // cur buffer fully staged; nxt buffer free to overwrite

        // Load fragments for 2x2 tiles from cur buffer
        v16bf ahi[2], alo[2], bhi[2], blo[2];
#pragma unroll
        for (int tm = 0; tm < 2; ++tm) {
            const int aoff = (32 * wm + 16 * tm + l16) * STRIDE + 8 * half;
            ahi[tm] = load_frag(&sAhi[cur][aoff], &sAhi[cur][aoff + 16]);
            alo[tm] = load_frag(&sAlo[cur][aoff], &sAlo[cur][aoff + 16]);
        }
#pragma unroll
        for (int tv = 0; tv < 2; ++tv) {
            const int boff = (32 * wv + 16 * tv + l16) * STRIDE + 8 * half;
            bhi[tv] = load_frag(&sBhi[cur][boff], &sBhi[cur][boff + 16]);
            blo[tv] = load_frag(&sBlo[cur][boff], &sBlo[cur][boff + 16]);
        }

        // bf16x3: fp32-accurate product on bf16 matrix cores (12 WMMA / wave / K-step)
#pragma unroll
        for (int tm = 0; tm < 2; ++tm) {
#pragma unroll
            for (int tv = 0; tv < 2; ++tv) {
                acc[tm][tv] = __builtin_amdgcn_wmma_f32_16x16x32_bf16(false, ahi[tm], false, bhi[tv], (short)0, acc[tm][tv], false, false);
                acc[tm][tv] = __builtin_amdgcn_wmma_f32_16x16x32_bf16(false, ahi[tm], false, blo[tv], (short)0, acc[tm][tv], false, false);
                acc[tm][tv] = __builtin_amdgcn_wmma_f32_16x16x32_bf16(false, alo[tm], false, bhi[tv], (short)0, acc[tm][tv], false, false);
            }
        }

        // Stage prefetched slab into the other buffer
        const int nxt = cur ^ 1;
        split_store4(pa0, &sAhi[nxt][ ldr       * STRIDE + ldc], &sAlo[nxt][ ldr       * STRIDE + ldc]);
        split_store4(pa1, &sAhi[nxt][(ldr + 32) * STRIDE + ldc], &sAlo[nxt][(ldr + 32) * STRIDE + ldc]);
        split_store4(pb0, &sBhi[nxt][ ldr       * STRIDE + ldc], &sBlo[nxt][ ldr       * STRIDE + ldc]);
        split_store4(pb1, &sBhi[nxt][(ldr + 32) * STRIDE + ldc], &sBlo[nxt][(ldr + 32) * STRIDE + ldc]);
        split_store4(pb2, &sBhi[nxt][(ldr + 64) * STRIDE + ldc], &sBlo[nxt][(ldr + 64) * STRIDE + ldc]);
        split_store4(pb3, &sBhi[nxt][(ldr + 96) * STRIDE + ldc], &sBlo[nxt][(ldr + 96) * STRIDE + ldc]);
        cur = nxt;
    }

    __syncthreads();

    // Epilogue: bias + softcap, per-row partial reductions.  Softcap bounds
    // |logit|<=30 so the exp-sum fits fp32 with no max-shift.
#pragma unroll
    for (int tv = 0; tv < 2; ++tv) {
        const int colg = v0 + 32 * wv + 16 * tv + l16;
        const float bv = bias[colg];
#pragma unroll
        for (int tm = 0; tm < 2; ++tm) {
#pragma unroll
            for (int r = 0; r < 8; ++r) {
                const int m = 32 * wm + 16 * tm + 8 * half + r;  // C layout: VGPR r -> M=r+8*half
                float lg = softcap30(acc[tm][tv][r] + bv);
                atomicAdd(&sExp[m], __expf(lg));
                atomicAdd(&sSum[m], lg);
                if (colg == sTgt[m]) g_zy[n0 + m] = lg;          // unique writer
            }
        }
    }
    __syncthreads();
    if (tid < BN) {
        atomicAdd(&g_sumexp[n0 + tid], sExp[tid]);
        atomicAdd(&g_sumlog[n0 + tid], sSum[tid]);
    }
}

__global__ __launch_bounds__(256)
void lce_finalize_kernel(const float* __restrict__ g_sumexp,
                         const float* __restrict__ g_sumlog,
                         const float* __restrict__ g_zy,
                         const int*   __restrict__ target,
                         float* __restrict__ out)
{
    __shared__ float ssum[256];
    __shared__ int   scnt[256];
    const int tid = threadIdx.x;
    float acc = 0.0f;
    int   cnt = 0;
    for (int n = tid; n < N_ROWS; n += 256) {
        int t = target[n];
        if (t != IGNORE_INDEX) {
            float lse  = logf(g_sumexp[n]);
            float loss = lse - 0.9f * g_zy[n]
                       - (0.1f / (float)V_DIM) * g_sumlog[n]
                       + 1e-4f * lse * lse;
            acc += loss;
            cnt += 1;
        }
    }
    ssum[tid] = acc;
    scnt[tid] = cnt;
    __syncthreads();
    for (int s = 128; s > 0; s >>= 1) {
        if (tid < s) { ssum[tid] += ssum[tid + s]; scnt[tid] += scnt[tid + s]; }
        __syncthreads();
    }
    if (tid == 0) {
        int nv = scnt[0] > 0 ? scnt[0] : 1;
        out[0] = ssum[0] / (float)nv;
    }
}

extern "C" void kernel_launch(void* const* d_in, const int* in_sizes, int n_in,
                              void* d_out, int out_size, void* d_ws, size_t ws_size,
                              hipStream_t stream) {
    const float* inp    = (const float*)d_in[0];   // [N,H]
    const int*   target = (const int*)  d_in[1];   // [N]
    const float* wgt    = (const float*)d_in[2];   // [V,H]
    const float* bias   = (const float*)d_in[3];   // [V]
    float* out = (float*)d_out;

    float* ws       = (float*)d_ws;
    float* g_sumexp = ws;
    float* g_sumlog = ws + N_ROWS;
    float* g_zy     = ws + 2 * N_ROWS;

    // zero the per-row accumulators every call (graph-capturable memset node)
    hipMemsetAsync(d_ws, 0, 3 * N_ROWS * sizeof(float), stream);

    dim3 grid(N_ROWS / BN, V_DIM / BV);   // (32, 250); x fastest => weight-slab L2 reuse
    lce_gemm_kernel<<<grid, 256, 0, stream>>>(inp, wgt, bias, target,
                                              g_sumexp, g_sumlog, g_zy);
    lce_finalize_kernel<<<1, 256, 0, stream>>>(g_sumexp, g_sumlog, g_zy, target, out);
}
